// S4ModNModelWithRetrospective_8418135900412
// MI455X (gfx1250) — compile-verified, matrix operand order checked by reading, not run on hardware
//
#include <hip/hip_runtime.h>
#include <hip/hip_bf16.h>
#include <math.h>

// ---- model dims ----
#define BSZ   256
#define SEQ   64
#define HDIM  512
#define NDIM  16
#define LNUM  2
#define TSTEP 63
#define ODIM  9

typedef __attribute__((ext_vector_type(16))) _Float16 v16h;
typedef __attribute__((ext_vector_type(8)))  _Float16 v8h;
typedef __attribute__((ext_vector_type(8)))  float    v8f;
typedef __attribute__((ext_vector_type(4)))  float    f4;

union FragH {
    v16h v;
    v8h  h8[2];
};

// ============================================================
// 1) Input projection: encoded = x = input_seq @ in_W + in_b
// ============================================================
__global__ void k_encode_proj(const float* __restrict__ in,   // (B*S,3)
                              const float* __restrict__ inW,  // (3,H)
                              const float* __restrict__ inb,  // (H)
                              float* __restrict__ encoded,    // (B*S,H)
                              float* __restrict__ x)          // (B*S,H)
{
    int idx = blockIdx.x * blockDim.x + threadIdx.x;   // over B*S*H
    int h  = idx & (HDIM - 1);
    int bs = idx >> 9;
    const float* r = in + bs * 3;
    float v = r[0] * inW[h] + r[1] * inW[HDIM + h] + r[2] * inW[2 * HDIM + h] + inb[h];
    encoded[idx] = v;
    x[idx] = v;
}

// ============================================================
// 2) Weight transpose + f16 convert: Wt[j][n][k] = outW[j][k][n]
// ============================================================
__global__ void k_convert_w(const float* __restrict__ outW,   // (L,H,H)
                            _Float16* __restrict__ Wt)        // (L,H,H) transposed
{
    int idx = blockIdx.x * blockDim.x + threadIdx.x;   // over L*H*H
    int k = idx & (HDIM - 1);
    int n = (idx >> 9) & (HDIM - 1);
    int j = idx >> 18;
    Wt[idx] = (_Float16)outW[j * HDIM * HDIM + k * HDIM + n];
}

// ============================================================
// fill / init
// ============================================================
__global__ void k_fill(float* __restrict__ p, int n, float v)
{
    int idx = blockIdx.x * blockDim.x + threadIdx.x;
    if (idx < n) p[idx] = v;
}

__global__ void k_init_decin(float* __restrict__ p)   // (B,3) = [0,0,1]
{
    int idx = blockIdx.x * blockDim.x + threadIdx.x;
    if (idx < BSZ * 3) p[idx] = ((idx % 3) == 2) ? 1.0f : 0.0f;
}

// ============================================================
// 3) S4 diagonal state update + gelu; writes f16 activation
// ============================================================
__global__ void k_s4_state(float* __restrict__ state,        // (B*H, N)
                           const float* __restrict__ xin,    // base (+rowStride per b)
                           int xRowStride,
                           const float* __restrict__ Aj,     // (H,N)
                           const float* __restrict__ Bj,     // (H,N)
                           const float* __restrict__ Cj,     // (H,N)
                           const float* __restrict__ Dj,     // (H)
                           _Float16* __restrict__ yact)      // (B*H) f16
{
    int idx = blockIdx.x * blockDim.x + threadIdx.x;   // b*H + h
    int h = idx & (HDIM - 1);
    int b = idx >> 9;
    float xv = xin[b * xRowStride + h];

    f4* sp = (f4*)(state + (size_t)idx * NDIM);
    const f4* ap = (const f4*)(Aj + h * NDIM);
    const f4* bp = (const f4*)(Bj + h * NDIM);
    const f4* cp = (const f4*)(Cj + h * NDIM);

    float acc = 0.0f;
#pragma unroll
    for (int q = 0; q < 4; ++q) {
        f4 a = ap[q], bm = bp[q], c = cp[q], s = sp[q];
        f4 s2 = a * s + bm * xv;
        sp[q] = s2;
        acc += s2[0] * c[0] + s2[1] * c[1] + s2[2] * c[2] + s2[3] * c[3];
    }
    float y = acc + Dj[h] * xv;
    // exact gelu
    float g = 0.5f * y * (1.0f + erff(y * 0.70710678118654752f));
    yact[idx] = (_Float16)g;
}

// ============================================================
// 4) WMMA GEMM: out(256,512) = yact(256,512)f16 @ W(512,512) + bias
//    8 waves/block share one 16-row A tile -> stage it in LDS via
//    async-load-to-LDS (double buffered), read back with ds_load_b128.
//    B fragments streamed per-wave from W^T (f16, row n = column of W).
// ============================================================
__global__ void k_wmma_gemm(const _Float16* __restrict__ A,   // (256,512) f16 row-major
                            const _Float16* __restrict__ Bt,  // (512,512) f16, W^T row-major
                            const float* __restrict__ bias,   // (512)
                            float* __restrict__ out)          // (256,512) f32
{
    __shared__ _Float16 tileA[2][16 * 32];   // double-buffered 16x32 A tile (1KB each)

    const int tid  = threadIdx.x;
    const int wave = tid >> 5;                  // 0..7
    const int lane = tid & 31;
    const int tileId = blockIdx.x * 8 + wave;   // 0..511
    const int tilesN = HDIM / 16;               // 32
    const int tm = (tileId / tilesN) * 16;      // identical for all 8 waves in block
    const int tn = (tileId % tilesN) * 16;
    const int laneHi = lane >> 4;               // K-half select
    const int lane15 = lane & 15;
    const int koff = laneHi * 8;                // halves

    // async staging: thread tid copies 4B of the 16x32 f16 A tile
    // row = tid/16, halves within row = (tid%16)*2
    const _Float16* aglob = A + (size_t)(tm + (tid >> 4)) * HDIM + (tid & 15) * 2;
    const unsigned lds0 = (unsigned)(size_t)(&tileA[0][0]) + tid * 4;
    const unsigned lds1 = (unsigned)(size_t)(&tileA[1][0]) + tid * 4;

    const _Float16* brow = Bt + (size_t)(tn + lane15) * HDIM;

    // prologue: stage k=0 into buffer 0
    asm volatile("global_load_async_to_lds_b32 %0, %1, off"
                 :: "v"(lds0), "v"(aglob) : "memory");
    asm volatile("s_wait_asynccnt 0" ::: "memory");
    __syncthreads();

    v8f acc = {};
    for (int k = 0; k < HDIM; k += 32) {
        const int buf = (k >> 5) & 1;
        if (k + 32 < HDIM) {
            // overlap: stage next k-chunk into the other buffer
            unsigned ldsn = buf ? lds0 : lds1;
            asm volatile("global_load_async_to_lds_b32 %0, %1, off"
                         :: "v"(ldsn), "v"(aglob + k + 32) : "memory");
        }
        // A fragment from LDS (16-bit A layout: lanes 0-15 K[k..k+7]&[k+16..k+23],
        // lanes 16-31 K[k+8..k+15]&[k+24..k+31]); row stride = 32 halves.
        FragH fa, fb;
        const _Float16* tA = &tileA[buf][0];
        fa.h8[0] = *(const v8h*)(tA + lane15 * 32 + koff);
        fa.h8[1] = *(const v8h*)(tA + lane15 * 32 + koff + 16);
        // B fragment from global (L2-resident weights)
        fb.h8[0] = *(const v8h*)(brow + k + koff);
        fb.h8[1] = *(const v8h*)(brow + k + koff + 16);
        if (k + 64 < HDIM) __builtin_prefetch(brow + k + 64, 0, 3);

        acc = __builtin_amdgcn_wmma_f32_16x16x32_f16(
            /*neg_a=*/false, fa.v, /*neg_b=*/false, fb.v,
            /*c_mod=*/(short)0, acc, /*reuse_a=*/false, /*reuse_b=*/false);

        if (k + 32 < HDIM) {
            asm volatile("s_wait_asynccnt 0" ::: "memory");
            __syncthreads();
        }
    }

    // C/D layout: VGPR v -> M = v + 8*laneHi, N = lane15
    const float bv = bias[tn + lane15];
#pragma unroll
    for (int v = 0; v < 8; ++v) {
        out[(size_t)(tm + v + 8 * laneHi) * HDIM + tn + lane15] = acc[v] + bv;
    }
}

// ============================================================
// 5) LayerNorm over H=512: out = LN(y + res) * gamma + beta
// ============================================================
__global__ void k_layernorm(const float* __restrict__ y,  int yStride,
                            const float* __restrict__ res, int resStride,
                            float* __restrict__ out,       int outStride,
                            const float* __restrict__ gamma,
                            const float* __restrict__ beta)
{
    int b = blockIdx.x, tid = threadIdx.x;
    __shared__ float red[256];
    float v0 = y[(size_t)b * yStride + tid]       + res[(size_t)b * resStride + tid];
    float v1 = y[(size_t)b * yStride + tid + 256] + res[(size_t)b * resStride + tid + 256];

    red[tid] = v0 + v1;
    __syncthreads();
    for (int s = 128; s > 0; s >>= 1) { if (tid < s) red[tid] += red[tid + s]; __syncthreads(); }
    float mean = red[0] * (1.0f / HDIM);
    __syncthreads();

    float d0 = v0 - mean, d1 = v1 - mean;
    red[tid] = d0 * d0 + d1 * d1;
    __syncthreads();
    for (int s = 128; s > 0; s >>= 1) { if (tid < s) red[tid] += red[tid + s]; __syncthreads(); }
    float rstd = rsqrtf(red[0] * (1.0f / HDIM) + 1e-5f);
    __syncthreads();

    out[(size_t)b * outStride + tid]       = d0 * rstd * gamma[tid]       + beta[tid];
    out[(size_t)b * outStride + tid + 256] = d1 * rstd * gamma[tid + 256] + beta[tid + 256];
}

// ============================================================
// 6) Decoder input projection: dp = dec_in(B,3) @ in_W + in_b
// ============================================================
__global__ void k_dec_proj(const float* __restrict__ dec_in, // (B,3)
                           const float* __restrict__ inW,    // (3,H)
                           const float* __restrict__ inb,    // (H)
                           float* __restrict__ dp)           // (B,H)
{
    int idx = blockIdx.x * blockDim.x + threadIdx.x;
    int h = idx & (HDIM - 1);
    int b = idx >> 9;
    const float* r = dec_in + b * 3;
    dp[idx] = r[0] * inW[h] + r[1] * inW[HDIM + h] + r[2] * inW[2 * HDIM + h] + inb[h];
}

// ============================================================
// 7) Head: sigmoid((xd+ctx)@hW+hb), pointer argmax, ctx update,
//    output write, next decoder token.  1 block / batch row.
// ============================================================
__device__ void head_dots(const float* xc, const float* __restrict__ hW,
                          const float* __restrict__ hb,
                          float* red, float* osig, int tid)
{
    float part[ODIM];
#pragma unroll
    for (int o = 0; o < ODIM; ++o) part[o] = 0.0f;
    for (int h = tid; h < HDIM; h += 256) {
        float xv = xc[h];
        const float* wr = hW + h * ODIM;
#pragma unroll
        for (int o = 0; o < ODIM; ++o) part[o] += xv * wr[o];
    }
    for (int o = 0; o < ODIM; ++o) {
        red[tid] = part[o];
        __syncthreads();
        for (int s = 128; s > 0; s >>= 1) { if (tid < s) red[tid] += red[tid + s]; __syncthreads(); }
        if (tid == 0) osig[o] = 1.0f / (1.0f + expf(-(red[0] + hb[o])));
        __syncthreads();
    }
}

__global__ void k_head(const float* __restrict__ xd,       // (B,H)
                       float* __restrict__ ctx,            // (B,H) updated in place
                       const float* __restrict__ encoded,  // (B,S,H)
                       const float* __restrict__ hW,       // (H,9)
                       const float* __restrict__ hb,       // (9)
                       float* __restrict__ out,            // (B,T,9)
                       float* __restrict__ dec_in,         // (B,3) next token
                       int t)
{
    int b = blockIdx.x, tid = threadIdx.x;
    __shared__ float xc[HDIM];
    __shared__ float red[256];
    __shared__ float osig[ODIM];
    __shared__ int   ptrS;

    xc[tid]       = xd[(size_t)b * HDIM + tid]       + ctx[(size_t)b * HDIM + tid];
    xc[tid + 256] = xd[(size_t)b * HDIM + tid + 256] + ctx[(size_t)b * HDIM + tid + 256];
    __syncthreads();

    head_dots(xc, hW, hb, red, osig, tid);

    if (tid == 0) {
        int pm = 0; float best = osig[3];
#pragma unroll
        for (int o = 1; o < 6; ++o) { if (osig[3 + o] > best) { best = osig[3 + o]; pm = o; } }
        ptrS = pm;
    }
    __syncthreads();

    const bool is_out = ((t & 1) == 0);
    if (!is_out) {
        // ctx += encoded[b, ptr, :]; recompute head with updated ctx
        const float* ev = encoded + ((size_t)b * SEQ + ptrS) * HDIM;
        float c0 = ctx[(size_t)b * HDIM + tid]       + ev[tid];
        float c1 = ctx[(size_t)b * HDIM + tid + 256] + ev[tid + 256];
        ctx[(size_t)b * HDIM + tid]       = c0;
        ctx[(size_t)b * HDIM + tid + 256] = c1;
        xc[tid]       = xd[(size_t)b * HDIM + tid]       + c0;
        xc[tid + 256] = xd[(size_t)b * HDIM + tid + 256] + c1;
        __syncthreads();
        head_dots(xc, hW, hb, red, osig, tid);
    }

    if (tid < ODIM) out[((size_t)b * TSTEP + t) * ODIM + tid] = osig[tid];
    if (tid == 0) {
        float nc = is_out ? ((osig[0] > 0.5f) ? 1.0f : 0.0f) : 0.0f;
        dec_in[b * 3 + 0] = nc;
        dec_in[b * 3 + 1] = is_out ? 1.0f : 0.0f;
        dec_in[b * 3 + 2] = is_out ? 0.0f : 1.0f;
    }
}

// ============================================================
// Host orchestration (graph-capturable: stream-only launches)
// ============================================================
extern "C" void kernel_launch(void* const* d_in, const int* in_sizes, int n_in,
                              void* d_out, int out_size, void* d_ws, size_t ws_size,
                              hipStream_t stream)
{
    const float* input_seq = (const float*)d_in[0];
    // d_in[1] = autoregressive_steps (scalar, unused: T fixed by OUT_BITS)
    const float* in_W   = (const float*)d_in[2];
    const float* in_b   = (const float*)d_in[3];
    const float* Amat   = (const float*)d_in[4];
    const float* Bmat   = (const float*)d_in[5];
    const float* Cmat   = (const float*)d_in[6];
    const float* Dvec   = (const float*)d_in[7];
    const float* outW   = (const float*)d_in[8];
    const float* outb   = (const float*)d_in[9];
    const float* ln_g   = (const float*)d_in[10];
    const float* ln_b   = (const float*)d_in[11];
    const float* head_W = (const float*)d_in[12];
    const float* head_b = (const float*)d_in[13];
    float* out = (float*)d_out;

    // ---- workspace carve-up (~92 MB, fits in 192 MB L2) ----
    char* ws = (char*)d_ws;
    float*    encoded   = (float*)ws;              ws += (size_t)BSZ * SEQ * HDIM * 4;
    float*    x         = (float*)ws;              ws += (size_t)BSZ * SEQ * HDIM * 4;
    float*    enc_state = (float*)ws;              ws += (size_t)BSZ * HDIM * NDIM * 4;
    float*    dec_state = (float*)ws;              ws += (size_t)LNUM * BSZ * HDIM * NDIM * 4;
    _Float16* Wt16      = (_Float16*)ws;           ws += (size_t)LNUM * HDIM * HDIM * 2;
    _Float16* yact      = (_Float16*)ws;           ws += (size_t)BSZ * HDIM * 2;
    float*    gemmout   = (float*)ws;              ws += (size_t)BSZ * HDIM * 4;
    float*    dp        = (float*)ws;              ws += (size_t)BSZ * HDIM * 4;
    float*    xd        = (float*)ws;              ws += (size_t)BSZ * HDIM * 4;
    float*    ctx       = (float*)ws;              ws += (size_t)BSZ * HDIM * 4;
    float*    dec_in    = (float*)ws;              ws += (size_t)BSZ * 3 * 4;

    const int HN = HDIM * NDIM;

    // ---- prologue ----
    k_encode_proj<<<(BSZ * SEQ * HDIM) / 256, 256, 0, stream>>>(input_seq, in_W, in_b, encoded, x);
    k_convert_w<<<(LNUM * HDIM * HDIM) / 256, 256, 0, stream>>>(outW, Wt16);

    // ---- encoder: L layers x S timesteps (sequential recurrence) ----
    for (int j = 0; j < LNUM; ++j) {
        k_fill<<<(BSZ * HDIM * NDIM) / 256, 256, 0, stream>>>(enc_state, BSZ * HDIM * NDIM, 0.0f);
        for (int t = 0; t < SEQ; ++t) {
            k_s4_state<<<(BSZ * HDIM) / 256, 256, 0, stream>>>(
                enc_state, x + (size_t)t * HDIM, SEQ * HDIM,
                Amat + j * HN, Bmat + j * HN, Cmat + j * HN, Dvec + j * HDIM, yact);
            k_wmma_gemm<<<64, 256, 0, stream>>>(yact, Wt16 + (size_t)j * HDIM * HDIM,
                                                outb + j * HDIM, gemmout);
            // x[:,t,:] = LN(ys_t + x[:,t,:]) in place
            k_layernorm<<<BSZ, 256, 0, stream>>>(gemmout, HDIM,
                                                 x + (size_t)t * HDIM, SEQ * HDIM,
                                                 x + (size_t)t * HDIM, SEQ * HDIM,
                                                 ln_g + j * HDIM, ln_b + j * HDIM);
        }
    }

    // ---- decoder init ----
    k_fill<<<(LNUM * BSZ * HDIM * NDIM) / 256, 256, 0, stream>>>(dec_state, LNUM * BSZ * HDIM * NDIM, 0.0f);
    k_fill<<<(BSZ * HDIM) / 256, 256, 0, stream>>>(ctx, BSZ * HDIM, 0.0f);
    k_init_decin<<<(BSZ * 3 + 255) / 256, 256, 0, stream>>>(dec_in);

    // ---- autoregressive decode: T sequential steps ----
    for (int t = 0; t < TSTEP; ++t) {
        k_dec_proj<<<(BSZ * HDIM) / 256, 256, 0, stream>>>(dec_in, in_W, in_b, dp);
        const float* layer_in = dp;
        for (int j = 0; j < LNUM; ++j) {
            k_s4_state<<<(BSZ * HDIM) / 256, 256, 0, stream>>>(
                dec_state + (size_t)j * BSZ * HN, layer_in, HDIM,
                Amat + j * HN, Bmat + j * HN, Cmat + j * HN, Dvec + j * HDIM, yact);
            k_wmma_gemm<<<64, 256, 0, stream>>>(yact, Wt16 + (size_t)j * HDIM * HDIM,
                                                outb + j * HDIM, gemmout);
            const float* res = (t == 0) ? dp : gemmout;  // step0: residual=proj, else 2*y
            k_layernorm<<<BSZ, 256, 0, stream>>>(gemmout, HDIM, res, HDIM, xd, HDIM,
                                                 ln_g + j * HDIM, ln_b + j * HDIM);
            layer_in = xd;
        }
        k_head<<<BSZ, 256, 0, stream>>>(xd, ctx, encoded, head_W, head_b, out, dec_in, t);
    }

    (void)in_sizes; (void)n_in; (void)out_size; (void)ws_size;
}
